// CFOCNet_86535001080247
// MI455X (gfx1250) — matrix operand order repaired and checked by reading, not compiled, not run on hardware
//
#include <hip/hip_runtime.h>

// ---------------------------------------------------------------------------
// CDNA5 (gfx1250) implementation of CFOCNet forward.
// Heavy GEMMs (attention projections, Q^T K, P V^T, match convs) run on
// v_wmma_f32_16x16x32_bf16 (fp32 accumulate). Flash-attention formulation
// avoids materializing the N x N attention matrix. All inner-loop trip counts
// are template constants so WMMA accumulators stay in place (no PHI copies,
// no scratch spills).
// ---------------------------------------------------------------------------

typedef unsigned short u16;
typedef unsigned int   u32;
typedef __attribute__((ext_vector_type(16))) __bf16 bf16x16;
typedef __attribute__((ext_vector_type(8)))  float  v8f;

union Frag16 { uint4 u[2]; bf16x16 v; };

__device__ __forceinline__ u16 f2bfu(float x) {
    u32 u = __builtin_bit_cast(u32, x);
    u32 r = (u + 0x7FFFu + ((u >> 16) & 1u)) >> 16;
    return (u16)r;
}
__device__ __forceinline__ float bf2f(u16 v) {
    u32 u = ((u32)v) << 16;
    return __builtin_bit_cast(float, u);
}

// A fragment (16x32 bf16) from row-major A; p = &A[row(lane&15)][k0].
// lane m elems 0..15 -> K=16g..16g+15 (contiguous, matches B fragment).
__device__ __forceinline__ bf16x16 load_a_frag_p(const u16* p, int g) {
    Frag16 f;
    f.u[0] = *(const uint4*)(p + 16 * g);
    f.u[1] = *(const uint4*)(p + 16 * g + 8);
    return f.v;
}
// B fragment (32x16 bf16) from column-contiguous storage; p already includes
// (col = lane&15) * ldb + k0 + 16g.
__device__ __forceinline__ bf16x16 load_b_frag_p(const u16* p) {
    Frag16 f;
    f.u[0] = *(const uint4*)(p);
    f.u[1] = *(const uint4*)(p + 8);
    return f.v;
}

__device__ __forceinline__ v8f wmma_bf16(bf16x16 a, bf16x16 b, v8f c) {
    return __builtin_amdgcn_wmma_f32_16x16x32_bf16(false, a, false, b, (short)0, c, false, false);
}

// ---------------------------------------------------------------------------
// Elementwise conversions
// ---------------------------------------------------------------------------
__global__ void cvt_bf16_k(const float* __restrict__ in, u16* __restrict__ out, long long n) {
    long long i = (long long)blockIdx.x * blockDim.x + threadIdx.x;
    if (i < n) out[i] = f2bfu(in[i]);
}

// x [B,C,N] f32 -> xt [B,N,C] bf16   (grid.z = B)
__global__ void cvt_transpose_k(const float* __restrict__ x, u16* __restrict__ xt, int C, int N) {
    long long i = (long long)blockIdx.x * blockDim.x + threadIdx.x;
    if (i >= (long long)C * N) return;
    int c = (int)(i / N), n = (int)(i % N);
    const float* xb = x + (size_t)blockIdx.z * C * N;
    u16* xtb = xt + (size_t)blockIdx.z * C * N;
    xtb[(size_t)n * C + c] = f2bfu(xb[i]);
}

// ---------------------------------------------------------------------------
// Batched bf16 WMMA GEMM:  C[m][n] = sum_k A[m][k]*Bt[n][k] + bias
// A row-major [M,K] (lda), Bt column-contiguous [Ncols,K] (ldb).
// Block = 128 threads = 4 waves; wave computes (MSUB*16) x (NSUB*16).
// Branch-free k-loop: accumulators stay pinned, WMMA updates in place.
// ---------------------------------------------------------------------------
template <int MSUB, int NSUB>
__global__ void gemm_bt_k(const u16* __restrict__ A, const u16* __restrict__ Bt,
                          const float* __restrict__ bias, void* __restrict__ Cout,
                          int M, int K, int lda, int ldb, int ldc,
                          long long sA, long long sB, long long sC,
                          int biasPerCol, int outBf16) {
    int b = blockIdx.z;
    int wave = threadIdx.x >> 5;
    int lane = threadIdx.x & 31;
    int g = lane >> 4, hl = lane & 15;
    int m0 = blockIdx.x * (MSUB * 16 * 4) + wave * (MSUB * 16);
    int n0 = blockIdx.y * (NSUB * 16);

    const u16* pa[MSUB];
#pragma unroll
    for (int mi = 0; mi < MSUB; ++mi)
        pa[mi] = A + (size_t)b * sA + (size_t)(m0 + mi * 16 + hl) * lda;
    const u16* pb[NSUB];
#pragma unroll
    for (int ni = 0; ni < NSUB; ++ni)
        pb[ni] = Bt + (size_t)b * sB + (size_t)(n0 + ni * 16 + hl) * ldb + 16 * g;

    v8f acc[MSUB][NSUB];
#pragma unroll
    for (int mi = 0; mi < MSUB; ++mi)
#pragma unroll
        for (int ni = 0; ni < NSUB; ++ni) acc[mi][ni] = (v8f){};

    for (int k0 = 0; k0 < K; k0 += 32) {
        bf16x16 a[MSUB];
#pragma unroll
        for (int mi = 0; mi < MSUB; ++mi) a[mi] = load_a_frag_p(pa[mi] + k0, g);
#pragma unroll
        for (int ni = 0; ni < NSUB; ++ni) {
            bf16x16 bb = load_b_frag_p(pb[ni] + k0);
#pragma unroll
            for (int mi = 0; mi < MSUB; ++mi)
                acc[mi][ni] = wmma_bf16(a[mi], bb, acc[mi][ni]);
        }
    }

#pragma unroll
    for (int mi = 0; mi < MSUB; ++mi) {
#pragma unroll
        for (int ni = 0; ni < NSUB; ++ni) {
            int n = n0 + ni * 16 + hl;
#pragma unroll
            for (int r = 0; r < 8; ++r) {
                int m = m0 + mi * 16 + r + 8 * g;
                float v = acc[mi][ni][r] + (biasPerCol ? bias[n] : bias[m]);
                if (outBf16)
                    ((u16*)Cout + (size_t)b * sC)[(size_t)m * ldc + n] = f2bfu(v);
                else
                    ((float*)Cout + (size_t)b * sC)[(size_t)m * ldc + n] = v;
            }
        }
    }
    (void)M;
}

// ---------------------------------------------------------------------------
// Flash self-attention: softmax(Q^T K) with streaming accumulation of P V^T.
// QT/KT [B,N,d] bf16, V [B,C,N] bf16, XT [B,N,C] bf16 -> YT [B,N,C] bf16.
// d = ND*32 (compile-time). grid = (N/64, C/128, B), block = 128.
// Wave: 16 queries x 128 channels, online softmax over all N keys.
// ---------------------------------------------------------------------------
template <int ND>
__global__ void flash_attn_k(const u16* __restrict__ QT, const u16* __restrict__ KT,
                             const u16* __restrict__ V, const u16* __restrict__ XT,
                             const float* __restrict__ gamma_p, u16* __restrict__ YT,
                             int N, int C) {
    const int d = ND * 32;
    __shared__ u16 plds[4][16 * 32];
    int b = blockIdx.z;
    int wave = threadIdx.x >> 5;
    int lane = threadIdx.x & 31;
    int g = lane >> 4, hl = lane & 15;
    int q0 = blockIdx.x * 64 + wave * 16;
    int c0 = blockIdx.y * 128;
    const u16* Qb = QT + (size_t)b * N * d;
    const u16* Kb = KT + (size_t)b * N * d;
    const u16* Vb = V + (size_t)b * C * N;
    const u16* Xb = XT + (size_t)b * N * C;
    u16* Yb = YT + (size_t)b * N * C;
    float gamma = gamma_p[0];

    float mrun[8], lrun[8];
#pragma unroll
    for (int r = 0; r < 8; ++r) { mrun[r] = -3.0e38f; lrun[r] = 0.f; }
    v8f o[8];
#pragma unroll
    for (int f = 0; f < 8; ++f) o[f] = (v8f){};

    bf16x16 qf[ND];
    {
        const u16* qp = Qb + (size_t)(q0 + hl) * d;
#pragma unroll
        for (int kc = 0; kc < ND; ++kc) qf[kc] = load_a_frag_p(qp + kc * 32, g);
    }
    const u16* kbase = Kb + (size_t)hl * d + 16 * g;   // + n0*d + kc*32
    const u16* vbase = Vb + (size_t)(c0 + hl) * N + 16 * g;  // + f*16*N + n0

    for (int n0 = 0; n0 < N; n0 += 32) {
        v8f s0 = (v8f){}, s1 = (v8f){};
#pragma unroll
        for (int kc = 0; kc < ND; ++kc) {
            bf16x16 b0 = load_b_frag_p(kbase + (size_t)n0 * d + kc * 32);
            bf16x16 b1 = load_b_frag_p(kbase + (size_t)(n0 + 16) * d + kc * 32);
            s0 = wmma_bf16(qf[kc], b0, s0);
            s1 = wmma_bf16(qf[kc], b1, s1);
        }
        // online softmax row stats (row = r + 8g, reduced over 16-lane half)
        float scl[8];
#pragma unroll
        for (int r = 0; r < 8; ++r) {
            float mx = fmaxf(s0[r], s1[r]);
#pragma unroll
            for (int off = 1; off < 16; off <<= 1) mx = fmaxf(mx, __shfl_xor(mx, off, 32));
            float mn = fmaxf(mrun[r], mx);
            scl[r] = __expf(mrun[r] - mn);
            mrun[r] = mn;
            float p0 = __expf(s0[r] - mn);
            float p1 = __expf(s1[r] - mn);
            s0[r] = p0; s1[r] = p1;
            float rs = p0 + p1;
#pragma unroll
            for (int off = 1; off < 16; off <<= 1) rs += __shfl_xor(rs, off, 32);
            lrun[r] = lrun[r] * scl[r] + rs;
        }
#pragma unroll
        for (int f = 0; f < 8; ++f)
#pragma unroll
            for (int r = 0; r < 8; ++r) o[f][r] *= scl[r];

        // P (C-layout) -> LDS -> A-fragment layout (same wave; DS stays in order)
        u16* pl = plds[wave];
#pragma unroll
        for (int r = 0; r < 8; ++r) {
            int row = r + 8 * g;
            pl[row * 32 + hl] = f2bfu(s0[r]);
            pl[row * 32 + 16 + hl] = f2bfu(s1[r]);
        }
        bf16x16 pf = load_a_frag_p(pl + hl * 32, g);

        // O += P * V^T-tile
#pragma unroll
        for (int f = 0; f < 8; ++f) {
            bf16x16 vf = load_b_frag_p(vbase + (size_t)(f * 16) * N + n0);
            o[f] = wmma_bf16(pf, vf, o[f]);
        }
    }
    // epilogue: Y^T[q][c] = gamma * O/l + X^T[q][c]
#pragma unroll
    for (int f = 0; f < 8; ++f) {
#pragma unroll
        for (int r = 0; r < 8; ++r) {
            int q = q0 + r + 8 * g;
            int c = c0 + 16 * f + hl;
            float xv = bf2f(Xb[(size_t)q * C + c]);
            float ov = o[f][r] / lrun[r];
            Yb[(size_t)q * C + c] = f2bfu(gamma * ov + xv);
        }
    }
}

// ---------------------------------------------------------------------------
// Reference branch: max over J then k x k spatial maxpool -> pooled [B,C,16]
// ---------------------------------------------------------------------------
__global__ void pool_ref_k(const float* __restrict__ r, float* __restrict__ pooled,
                           int J, int C, int S, int k) {
    int idx = blockIdx.x * blockDim.x + threadIdx.x;
    if (idx >= C * 16) return;
    int b = blockIdx.z;
    int c = idx >> 4, s = idx & 15;
    int sy = s >> 2, sx = s & 3;
    float m = -3.0e38f;
    for (int j = 0; j < J; ++j) {
        const float* rb = r + (((size_t)b * J + j) * C + c) * S * S;
        for (int dy = 0; dy < k; ++dy)
            for (int dx = 0; dx < k; ++dx)
                m = fmaxf(m, rb[(sy * k + dy) * S + sx * k + dx]);
    }
    pooled[((size_t)b * C + c) * 16 + s] = m;
}

// kref [B,256,16] = mr_w [256,C] @ pooled [B,C,16] + mr_b
__global__ void ref_conv_k(const float* __restrict__ pooled, const float* __restrict__ w,
                           const float* __restrict__ bias, float* __restrict__ kout, int C) {
    int idx = blockIdx.x * blockDim.x + threadIdx.x;
    if (idx >= 256 * 16) return;
    int b = blockIdx.z;
    int o = idx >> 4, s = idx & 15;
    const float* pb = pooled + (size_t)b * C * 16;
    float acc = bias[o];
    for (int c = 0; c < C; ++c) acc += w[(size_t)o * C + c] * pb[c * 16 + s];
    kout[((size_t)b * 256 + o) * 16 + s] = acc;
}

// ---------------------------------------------------------------------------
// Dynamic-kernel correlation: in [B,256,H,H] (x) kref [B,256,4,4], stride 2,
// pad 1 -> M [B,h,h].  One wave per output pixel.
// ---------------------------------------------------------------------------
__global__ void corr_k(const float* __restrict__ in, const float* __restrict__ ker,
                       float* __restrict__ M, int H, int h) {
    int pix = blockIdx.x;
    int lane = threadIdx.x;
    int b = pix / (h * h);
    int rem = pix % (h * h);
    int y = rem / h, x = rem % h;
    const float* inb = in + (size_t)b * 256 * H * H;
    const float* kb = ker + (size_t)b * 256 * 16;
    float acc = 0.f;
    for (int c = lane; c < 256; c += 32) {
        const float* ic = inb + (size_t)c * H * H;
        const float* kc = kb + c * 16;
        for (int ky = 0; ky < 4; ++ky) {
            int iy = 2 * y - 1 + ky;
            if (iy < 0 || iy >= H) continue;
            for (int kx = 0; kx < 4; ++kx) {
                int ix = 2 * x - 1 + kx;
                if (ix < 0 || ix >= H) continue;
                acc += ic[iy * H + ix] * kc[ky * 4 + kx];
            }
        }
    }
    for (int off = 1; off < 32; off <<= 1) acc += __shfl_xor(acc, off, 32);
    if (lane == 0) M[pix] = acc;
}

// ---------------------------------------------------------------------------
// Per-batch scale weights: softmax over [s_i_w * sum(M_i_up) + 1024 * s_i_b]
// ---------------------------------------------------------------------------
__global__ void scale_softmax_k(const float* __restrict__ M1, const float* __restrict__ M2,
                                const float* __restrict__ M3,
                                const float* s1w, const float* s1b,
                                const float* s2w, const float* s2b,
                                const float* s3w, const float* s3b,
                                float* __restrict__ Wf) {
    __shared__ float red[128];
    int b = blockIdx.x, t = threadIdx.x;
    float sums[3] = {0.f, 0.f, 0.f};
    for (int i = t; i < 1024; i += 128) sums[0] += M1[b * 1024 + i];
    for (int i = t; i < 256; i += 128) sums[1] += M2[b * 256 + i];
    for (int i = t; i < 64; i += 128) sums[2] += M3[b * 64 + i];
    float tot[3];
    for (int k = 0; k < 3; ++k) {
        red[t] = sums[k];
        __syncthreads();
        for (int off = 64; off > 0; off >>= 1) {
            if (t < off) red[t] += red[t + off];
            __syncthreads();
        }
        tot[k] = red[0];
        __syncthreads();
    }
    if (t == 0) {
        float z0 = s1w[0] * tot[0] + 1024.f * s1b[0];
        float z1 = s2w[0] * 4.f * tot[1] + 1024.f * s2b[0];
        float z2 = s3w[0] * 16.f * tot[2] + 1024.f * s3b[0];
        float mx = fmaxf(z0, fmaxf(z1, z2));
        float e0 = __expf(z0 - mx), e1 = __expf(z1 - mx), e2 = __expf(z2 - mx);
        float inv = 1.f / (e0 + e1 + e2);
        Wf[b * 3 + 0] = e0 * inv;
        Wf[b * 3 + 1] = e1 * inv;
        Wf[b * 3 + 2] = e2 * inv;
    }
}

// FS32[b,y,x] = W0*M1 + W1*M2up + W2*M3up (nearest upsample via index shift)
__global__ void fs32_k(const float* __restrict__ M1, const float* __restrict__ M2,
                       const float* __restrict__ M3, const float* __restrict__ Wf,
                       float* __restrict__ FS) {
    int i = blockIdx.x * blockDim.x + threadIdx.x;
    if (i >= 4096) return;
    int b = i >> 10, rem = i & 1023;
    int y = rem >> 5, x = rem & 31;
    FS[i] = Wf[b * 3 + 0] * M1[i]
          + Wf[b * 3 + 1] * M2[b * 256 + (y >> 1) * 16 + (x >> 1)]
          + Wf[b * 3 + 2] * M3[b * 64 + (y >> 2) * 8 + (x >> 2)];
}

// ConvTranspose2d(1,1,3,s=2,p=1,op=1): FS64[y,x] from lhs-dilated conv
__global__ void fs64_k(const float* __restrict__ FS32, const float* __restrict__ tcw,
                       const float* __restrict__ tcb, float* __restrict__ FS64) {
    int i = blockIdx.x * blockDim.x + threadIdx.x;
    if (i >= 16384) return;
    int b = i >> 12, rem = i & 4095;
    int y = rem >> 6, x = rem & 63;
    float acc = tcb[0];
    for (int a = 0; a < 3; ++a) {
        int u = y + a - 1;
        if (u < 0 || u > 62 || (u & 1)) continue;
        for (int c = 0; c < 3; ++c) {
            int v = x + c - 1;
            if (v < 0 || v > 62 || (v & 1)) continue;
            acc += FS32[b * 1024 + (u >> 1) * 32 + (v >> 1)] * tcw[(2 - a) * 3 + (2 - c)];
        }
    }
    FS64[i] = acc;
}

// Bilinear x4, align_corners=True: 64 -> 256
__global__ void bilin_k(const float* __restrict__ FS64, float* __restrict__ out) {
    int i = blockIdx.x * blockDim.x + threadIdx.x;
    if (i >= 262144) return;
    int b = i >> 16, rem = i & 65535;
    int oy = rem >> 8, ox = rem & 255;
    float ys = oy * (63.f / 255.f);
    float xs = ox * (63.f / 255.f);
    int y0 = (int)floorf(ys); int y1 = min(y0 + 1, 63);
    int x0 = (int)floorf(xs); int x1 = min(x0 + 1, 63);
    float wy = ys - (float)y0, wx = xs - (float)x0;
    const float* F = FS64 + b * 4096;
    float v00 = F[y0 * 64 + x0], v01 = F[y0 * 64 + x1];
    float v10 = F[y1 * 64 + x0], v11 = F[y1 * 64 + x1];
    out[i] = (v00 * (1.f - wy) + v10 * wy) * (1.f - wx)
           + (v01 * (1.f - wy) + v11 * wy) * wx;
}

// ---------------------------------------------------------------------------
// Host-side orchestration
// ---------------------------------------------------------------------------
static void launch_gemm(const u16* A, const u16* Bt, const float* bias, void* Cout,
                        int M, int Ncols, int K, int lda, int ldb, int ldc,
                        long long sA, long long sB, long long sC,
                        int biasPerCol, int outBf16, int batch, hipStream_t stream) {
    // M always divisible by 128; Ncols is 32 or a multiple of 64.
    if (Ncols == 32) {
        dim3 grid(M / 128, 1, batch);
        gemm_bt_k<2, 2><<<grid, dim3(128), 0, stream>>>(A, Bt, bias, Cout, M, K,
            lda, ldb, ldc, sA, sB, sC, biasPerCol, outBf16);
    } else {
        dim3 grid(M / 128, Ncols / 64, batch);
        gemm_bt_k<2, 4><<<grid, dim3(128), 0, stream>>>(A, Bt, bias, Cout, M, K,
            lda, ldb, ldc, sA, sB, sC, biasPerCol, outBf16);
    }
}

extern "C" void kernel_launch(void* const* d_in, const int* in_sizes, int n_in,
                              void* d_out, int out_size, void* d_ws, size_t ws_size,
                              hipStream_t stream) {
    (void)in_sizes; (void)n_in; (void)out_size; (void)ws_size;
    const int B = 4, J = 5;
    const int Cs[3] = {256, 512, 1024};
    const int Ns[3] = {4096, 1024, 256};
    const int Hs[3] = {64, 32, 16};
    const int hs_[3] = {32, 16, 8};
    const int Ss[3] = {16, 8, 4};
    const int pks[3] = {4, 2, 1};

    // workspace arena (fixed offsets, 256B aligned)
    char* wptr = (char*)d_ws;
    auto alloc = [&](size_t bytes) -> char* {
        char* p = wptr;
        wptr += (bytes + 255) & ~(size_t)255;
        return p;
    };
    u16* XT = (u16*)alloc((size_t)B * 4096 * 256 * 2);   // max N*C elems
    u16* QT = (u16*)alloc((size_t)B * 4096 * 32 * 2);
    u16* KT = (u16*)alloc((size_t)B * 4096 * 32 * 2);
    u16* Vb = (u16*)alloc((size_t)B * 256 * 4096 * 2);
    u16* YT = (u16*)alloc((size_t)B * 4096 * 256 * 2);
    float* inb = (float*)alloc((size_t)B * 256 * 4096 * 4);
    u16* Wqb = (u16*)alloc((size_t)128 * 1024 * 2);
    u16* Wkb = (u16*)alloc((size_t)128 * 1024 * 2);
    u16* Wvb = (u16*)alloc((size_t)1024 * 1024 * 2);
    u16* Mqb = (u16*)alloc((size_t)256 * 1024 * 2);
    float* pooled = (float*)alloc((size_t)B * 1024 * 16 * 4);
    float* kref = (float*)alloc((size_t)B * 256 * 16 * 4);
    float* Mbuf0 = (float*)alloc((size_t)B * 1024 * 4);
    float* Mbuf1 = (float*)alloc((size_t)B * 256 * 4);
    float* Mbuf2 = (float*)alloc((size_t)B * 64 * 4);
    float* Wf = (float*)alloc(256);
    float* FS32 = (float*)alloc((size_t)B * 1024 * 4);
    float* FS64 = (float*)alloc((size_t)B * 4096 * 4);
    float* Mbufs[3] = {Mbuf0, Mbuf1, Mbuf2};

    for (int s = 0; s < 3; ++s) {
        const int C = Cs[s], d = C / 8, N = Ns[s], H = Hs[s], h = hs_[s];
        const float* x = (const float*)d_in[s];
        const float* r = (const float*)d_in[3 + s];
        int base = 6 + s * 13;
        const float* wq  = (const float*)d_in[base + 0];
        const float* bq  = (const float*)d_in[base + 1];
        const float* wk  = (const float*)d_in[base + 2];
        const float* bk  = (const float*)d_in[base + 3];
        const float* wv  = (const float*)d_in[base + 4];
        const float* bv  = (const float*)d_in[base + 5];
        const float* ga  = (const float*)d_in[base + 6];
        const float* mqw = (const float*)d_in[base + 7];
        const float* mqb = (const float*)d_in[base + 8];
        const float* mrw = (const float*)d_in[base + 9];
        const float* mrb = (const float*)d_in[base + 10];

        // bf16 weight conversions
        {
            long long n;
            n = (long long)d * C;   cvt_bf16_k<<<dim3((unsigned)((n + 255) / 256)), 256, 0, stream>>>(wq, Wqb, n);
            n = (long long)d * C;   cvt_bf16_k<<<dim3((unsigned)((n + 255) / 256)), 256, 0, stream>>>(wk, Wkb, n);
            n = (long long)C * C;   cvt_bf16_k<<<dim3((unsigned)((n + 255) / 256)), 256, 0, stream>>>(wv, Wvb, n);
            n = (long long)256 * C; cvt_bf16_k<<<dim3((unsigned)((n + 255) / 256)), 256, 0, stream>>>(mqw, Mqb, n);
        }
        // X [B,C,N] -> XT [B,N,C] bf16
        {
            long long n = (long long)C * N;
            cvt_transpose_k<<<dim3((unsigned)((n + 255) / 256), 1, B), 256, 0, stream>>>(x, XT, C, N);
        }
        // Q^T [N,d] = XT @ Wq^T + bq  (bias per column), bf16 out
        launch_gemm(XT, Wqb, bq, QT, N, d, C, C, C, d,
                    (long long)N * C, 0, (long long)N * d, 1, 1, B, stream);
        // K^T [N,d]
        launch_gemm(XT, Wkb, bk, KT, N, d, C, C, C, d,
                    (long long)N * C, 0, (long long)N * d, 1, 1, B, stream);
        // V [C,N] = Wv @ X + bv  (bias per row), bf16 out
        launch_gemm(Wvb, XT, bv, Vb, C, N, C, C, C, N,
                    0, (long long)N * C, (long long)C * N, 0, 1, B, stream);
        // flash attention -> YT [B,N,C] bf16 (= gamma*O + X, transposed layout)
        {
            dim3 fgrid(N / 64, C / 128, B);
            if (d == 32)
                flash_attn_k<1><<<fgrid, dim3(128), 0, stream>>>(QT, KT, Vb, XT, ga, YT, N, C);
            else if (d == 64)
                flash_attn_k<2><<<fgrid, dim3(128), 0, stream>>>(QT, KT, Vb, XT, ga, YT, N, C);
            else
                flash_attn_k<4><<<fgrid, dim3(128), 0, stream>>>(QT, KT, Vb, XT, ga, YT, N, C);
        }
        // match conv: in [B,256,N] f32 = Mq @ Y + mq_b
        launch_gemm(Mqb, YT, mqb, inb, 256, N, C, C, C, N,
                    0, (long long)N * C, (long long)256 * N, 0, 0, B, stream);
        // reference branch
        pool_ref_k<<<dim3((C * 16 + 255) / 256, 1, B), 256, 0, stream>>>(r, pooled, J, C, Ss[s], pks[s]);
        ref_conv_k<<<dim3((4096 + 255) / 256, 1, B), 256, 0, stream>>>(pooled, mrw, mrb, kref, C);
        // correlation -> M_s [B,h,h]
        corr_k<<<dim3(B * h * h), dim3(32), 0, stream>>>(inb, kref, Mbufs[s], H, h);
    }

    const float* s1w = (const float*)d_in[17]; const float* s1b = (const float*)d_in[18];
    const float* s2w = (const float*)d_in[30]; const float* s2b = (const float*)d_in[31];
    const float* s3w = (const float*)d_in[43]; const float* s3b = (const float*)d_in[44];
    const float* tcw = (const float*)d_in[45]; const float* tcb = (const float*)d_in[46];

    scale_softmax_k<<<dim3(B), dim3(128), 0, stream>>>(Mbuf0, Mbuf1, Mbuf2,
                                                       s1w, s1b, s2w, s2b, s3w, s3b, Wf);
    fs32_k<<<dim3(16), dim3(256), 0, stream>>>(Mbuf0, Mbuf1, Mbuf2, Wf, FS32);
    fs64_k<<<dim3(64), dim3(256), 0, stream>>>(FS32, tcw, tcb, FS64);
    bilin_k<<<dim3(1024), dim3(256), 0, stream>>>(FS64, (float*)d_out);
}
